// MultiHeadAttention_670014898657
// MI455X (gfx1250) — matrix operand order repaired and checked by reading, not compiled, not run on hardware
//
#include <hip/hip_runtime.h>
#include <hip/hip_bf16.h>
#include <math.h>

#define DMODEL 768
#define NHEADS 12
#define DK     64
#define SEQ    2048
#define BATCH  4
#define BS     (BATCH*SEQ)

typedef __attribute__((ext_vector_type(16))) _Float16 v16h;
typedef __attribute__((ext_vector_type(8)))  _Float16 v8h;
typedef __attribute__((ext_vector_type(8)))  float    v8f;
typedef __attribute__((ext_vector_type(4)))  unsigned int u32x4;
typedef __attribute__((ext_vector_type(8)))  int          i32x8;
typedef __attribute__((ext_vector_type(4)))  int          i32x4;

#if __has_builtin(__builtin_amdgcn_tensor_load_to_lds) && \
    __has_builtin(__builtin_amdgcn_s_wait_tensorcnt)
#define HAVE_TDM 1
#else
#define HAVE_TDM 0
#endif

static __device__ inline v8f vzero8() {
  v8f z;
#pragma unroll
  for (int i = 0; i < 8; ++i) z[i] = 0.0f;
  return z;
}

static __device__ inline v8f wmma_f16(v16h a, v16h b, v8f c) {
  return __builtin_amdgcn_wmma_f32_16x16x32_f16(false, a, false, b, (short)0, c,
                                                false, false);
}

// Low 32 bits of a generic pointer to __shared__ = LDS byte offset (flat aperture).
static __device__ inline unsigned lds_off(const void* p) {
  return (unsigned)(size_t)p;
}

#if HAVE_TDM
// TDM 2D f16 tile load: cols x rows tile, global row stride gstride (elements).
static __device__ inline void tdm_load_tile_f16(const _Float16* gsrc,
                                                unsigned ldsb, unsigned cols,
                                                unsigned rows, unsigned gstride) {
  unsigned long long ga = (unsigned long long)(size_t)gsrc;
  u32x4 g0;
  g0[0] = 1u;                                           // count=1 (valid user D#)
  g0[1] = ldsb;                                         // lds_addr (bytes)
  g0[2] = (unsigned)ga;                                 // global_addr[31:0]
  g0[3] = ((unsigned)(ga >> 32) & 0x01FFFFFFu) | 0x80000000u; // addr[56:32]|type=2
  i32x8 g1;
  g1[0] = (int)(1u << 16);                              // data_size = 2 bytes
  g1[1] = (int)(cols << 16);                            // tensor_dim0[15:0]
  g1[2] = (int)((cols >> 16) | (rows << 16));           // dim0 hi | tensor_dim1 lo
  g1[3] = (int)((rows >> 16) | (cols << 16));           // dim1 hi | tile_dim0
  g1[4] = (int)(rows & 0xFFFFu);                        // tile_dim1 (tile_dim2=0)
  g1[5] = (int)gstride;                                 // tensor_dim0_stride lo
  g1[6] = 0;
  g1[7] = 0;
  i32x4 z4;
  z4[0] = 0; z4[1] = 0; z4[2] = 0; z4[3] = 0;
#if defined(__clang_major__) && (__clang_major__ >= 23)
  i32x8 z8;
#pragma unroll
  for (int i = 0; i < 8; ++i) z8[i] = 0;
  __builtin_amdgcn_tensor_load_to_lds(g0, g1, z4, z4, z8, 0);
#else
  __builtin_amdgcn_tensor_load_to_lds(g0, g1, z4, z4, 0);
#endif
}
#endif

// ---- cross-lane max over the 16-lane row via DPP rotate-reduce ----
#if __has_builtin(__builtin_amdgcn_update_dpp)
template <int N>
static __device__ inline float dpp_ror16(float x) {
  return __int_as_float(__builtin_amdgcn_update_dpp(
      0, __float_as_int(x), 0x120 | N, 0xF, 0xF, true));
}
static __device__ inline float redmax16(float x) {
  x = fmaxf(x, dpp_ror16<8>(x));
  x = fmaxf(x, dpp_ror16<4>(x));
  x = fmaxf(x, dpp_ror16<2>(x));
  x = fmaxf(x, dpp_ror16<1>(x));
  return x;
}
#else
static __device__ inline float redmax16(float x) {
#pragma unroll
  for (int off = 8; off >= 1; off >>= 1) x = fmaxf(x, __shfl_xor(x, off, 32));
  return x;
}
#endif

// ---- CDNA5 WMMA fragment loaders (wave32 layouts, ISA 7.12.2) ----
// A fragment 16x32 (MxK), row-major, base = &A[row0*ld + k0]. Contiguous.
static __device__ inline v16h frag_a(const _Float16* base, int ld) {
  const int l  = threadIdx.x & 31;
  const int m  = l & 15;
  const int kb = (l >> 4) << 3;
  const _Float16* p = base + m * ld + kb;
  union { v16h v; v8h h[2]; } u;
  u.h[0] = *(const v8h*)(p);
  u.h[1] = *(const v8h*)(p + 16);
  return u.v;
}

// B fragment 32x16 (KxN) from N-major memory: element (kk,n) at base[n*ld+kk].
// Per lane: k = 8h..8h+7 and 16+8h..16+8h+7 (two b128 LDS loads), matching frag_a.
static __device__ inline v16h frag_b_nmajor(const _Float16* base, int ld) {
  const int l = threadIdx.x & 31;
  const _Float16* p = base + (size_t)(l & 15) * ld + ((l >> 4) << 3);
  union { v16h v; v8h h[2]; } u;
  u.h[0] = *(const v8h*)(p);
  u.h[1] = *(const v8h*)(p + 16);
  return u.v;
}

// ------------------- f32 -> f16 conversion -------------------
__global__ __launch_bounds__(256) void cvt_f16_kernel(const float* __restrict__ in,
                                                      _Float16* __restrict__ out,
                                                      int n) {
  int i = blockIdx.x * 256 + threadIdx.x;
  if (i < n) out[i] = (_Float16)in[i];
}

// f32 [D x D] -> f16 transposed [D x D] (output write-coalesced, one-time)
__global__ __launch_bounds__(256) void cvt_f16_transpose_kernel(
    const float* __restrict__ in, _Float16* __restrict__ out, int D) {
  int i = blockIdx.x * 256 + threadIdx.x;
  if (i < D * D) {
    int n = i / D, k = i - n * D;
    out[i] = (_Float16)in[(size_t)k * D + n];
  }
}

// f16 per-head transpose: vh[b,s,h*64+d] -> vt[b,h,d,s] (LDS-tiled, coalesced)
__global__ __launch_bounds__(256) void transpose_v_kernel(
    const _Float16* __restrict__ vh, _Float16* __restrict__ vt) {
  __shared__ _Float16 t[64][65];
  const int s0 = blockIdx.x * 64;
  const int h  = blockIdx.y;
  const int b  = blockIdx.z;
  const int tid = threadIdx.x;
#pragma unroll
  for (int tt = 0; tt < 16; ++tt) {
    int idx = tt * 256 + tid;
    int sr = idx >> 6, d = idx & 63;
    t[sr][d] = vh[((size_t)b * SEQ + s0 + sr) * DMODEL + h * DK + d];
  }
  __syncthreads();
#pragma unroll
  for (int tt = 0; tt < 16; ++tt) {
    int idx = tt * 256 + tid;
    int d = idx >> 6, sr = idx & 63;
    vt[(((size_t)b * NHEADS + h) * DK + d) * SEQ + s0 + sr] = t[sr][d];
  }
}

// ------------------- WMMA GEMM: C = A * W + bias -------------------
// A: f16 row-major [M x K]. WhT: f16 N-major [N x K] (pre-transposed weights).
// 256 threads = 8 waves, 128x64 tile. Both LDS tiles are straight 2D TDM
// copies, double-buffered; s_wait_tensorcnt 2 keeps next iteration's DMAs
// in flight under the WMMAs. All fragment reads are contiguous b128 LDS loads.
template <int EPI>
__global__ __launch_bounds__(256) void gemm_wmma_kernel(
    const _Float16* __restrict__ Ah, const _Float16* __restrict__ WhT,
    const float* __restrict__ bias, const float* __restrict__ residual,
    _Float16* __restrict__ outh, float* __restrict__ outf,
    int M, int N, int K) {
  __shared__ __align__(16) _Float16 As[2][128 * 32];   // [m][k] ld=32
  __shared__ __align__(16) _Float16 Wt[2][64 * 32];    // [n][k] ld=32
  const int tid  = threadIdx.x;
  const int wid  = tid >> 5;
  const int lane = tid & 31;
  const int n0 = blockIdx.x * 64;
  const int m0 = blockIdx.y * 128;

  v8f acc[4];
#pragma unroll
  for (int j = 0; j < 4; ++j) acc[j] = vzero8();

#if HAVE_TDM
  if (wid == 0) {   // prologue: tile 0 DMAs
    tdm_load_tile_f16(Ah + (size_t)m0 * K, lds_off(&As[0][0]), 32, 128, K);
    tdm_load_tile_f16(WhT + (size_t)n0 * K, lds_off(&Wt[0][0]), 32, 64, K);
  }
#endif

  const int NIT = K / 32;
  for (int it = 0; it < NIT; ++it) {
    const int k0  = it * 32;
    const int cur = it & 1;
    const int nxt = cur ^ 1;
    const bool next = (it + 1 < NIT);
#if HAVE_TDM
    if (wid == 0) {
      if (next) {
        tdm_load_tile_f16(Ah + (size_t)m0 * K + k0 + 32, lds_off(&As[nxt][0]),
                          32, 128, K);
        tdm_load_tile_f16(WhT + (size_t)n0 * K + k0 + 32, lds_off(&Wt[nxt][0]),
                          32, 64, K);
        __builtin_amdgcn_s_wait_tensorcnt(2);  // drain tile it, keep it+1 flying
      } else {
        __builtin_amdgcn_s_wait_tensorcnt(0);
      }
    }
#else
#pragma unroll
    for (int t = 0; t < 16; ++t) {
      int idx = t * 256 + tid;
      int r = idx >> 5, c = idx & 31;
      As[cur][r * 32 + c] = Ah[(size_t)(m0 + r) * K + k0 + c];
    }
#pragma unroll
    for (int t = 0; t < 8; ++t) {
      int idx = t * 256 + tid;
      int r = idx >> 5, c = idx & 31;
      Wt[cur][r * 32 + c] = WhT[(size_t)(n0 + r) * K + k0 + c];
    }
#endif
    __syncthreads();

    v16h a = frag_a(&As[cur][0] + wid * 16 * 32, 32);
#pragma unroll
    for (int j = 0; j < 4; ++j) {
      v16h b = frag_b_nmajor(&Wt[cur][0] + (j * 16) * 32, 32);
      acc[j] = wmma_f16(a, b, acc[j]);
    }
    __syncthreads();
  }

  const int hh = lane >> 4, n = lane & 15;
#pragma unroll
  for (int j = 0; j < 4; ++j) {
#pragma unroll
    for (int r = 0; r < 8; ++r) {
      int row = m0 + wid * 16 + hh * 8 + r;
      int col = n0 + j * 16 + n;
      float v = acc[j][r] + bias[col];
      if constexpr (EPI == 0) {
        outh[(size_t)row * N + col] = (_Float16)v;
      } else {
        outf[(size_t)row * N + col] = v + residual[(size_t)row * N + col];
      }
    }
  }
}

// ------------------- flash attention -------------------
// grid (SEQ/64, NHEADS, BATCH); 128 thr = 4 waves; wave owns 16 query rows.
// K chunk [kv][d] and V^T chunk [d][kv] (from pre-transposed vt) are straight
// 2D TDM copies, double-buffered with s_wait_tensorcnt 2. All fragments are
// contiguous b128 LDS loads. Row-max via DPP, row-sum via WMMA vs ones.
__global__ __launch_bounds__(128) void attn_kernel(
    const _Float16* __restrict__ qb, const _Float16* __restrict__ kb,
    const _Float16* __restrict__ vt, const unsigned char* __restrict__ mask,
    _Float16* __restrict__ ctx) {
  __shared__ __align__(16) _Float16 Kl[2][32 * 64];    // [kv][d] ld=64
  __shared__ __align__(16) _Float16 Vt[2][64 * 32];    // [d][kv] ld=32
  __shared__ __align__(16) _Float16 Plds[4][16 * 32];  // per-wave P tile
  const int tid  = threadIdx.x;
  const int wid  = tid >> 5;
  const int lane = tid & 31;
  const int h  = blockIdx.y;
  const int b  = blockIdx.z;
  const int q0 = blockIdx.x * 64 + wid * 16;
  const int hh = lane >> 4, n = lane & 15;

  const size_t qbase = ((size_t)b * SEQ + q0) * DMODEL + h * DK;
  v16h aq0 = frag_a(qb + qbase + 0,  DMODEL);
  v16h aq1 = frag_a(qb + qbase + 32, DMODEL);
#pragma unroll
  for (int e = 0; e < 16; ++e) {   // fold 1/sqrt(DK) into Q once
    aq0[e] = aq0[e] * (_Float16)0.125f;
    aq1[e] = aq1[e] * (_Float16)0.125f;
  }

  v16h ones;
#pragma unroll
  for (int e = 0; e < 16; ++e) ones[e] = (_Float16)1.0f;

  v8f O[4];
#pragma unroll
  for (int j = 0; j < 4; ++j) O[j] = vzero8();
  float m[8], l[8];
#pragma unroll
  for (int r = 0; r < 8; ++r) { m[r] = -3.0e38f; l[r] = 0.0f; }

  const size_t mb     = (size_t)b * SEQ * SEQ;
  const size_t kbase  = (size_t)b * SEQ * DMODEL + h * DK;          // K rows
  const size_t vtbase = ((size_t)b * NHEADS + h) * DK * SEQ;        // V^T rows

#if HAVE_TDM
  if (wid == 0) {   // prologue: chunk 0 DMAs
    tdm_load_tile_f16(kb + kbase, lds_off(&Kl[0][0]), 64, 32, DMODEL);
    tdm_load_tile_f16(vt + vtbase, lds_off(&Vt[0][0]), 32, 64, SEQ);
  }
#endif

  for (int kv0 = 0; kv0 < SEQ; kv0 += 32) {
    const int it  = kv0 >> 5;
    const int cur = it & 1;
    const int nxt = cur ^ 1;
    const bool next = (kv0 + 32 < SEQ);
#if HAVE_TDM
    if (wid == 0) {
      if (next) {
        tdm_load_tile_f16(kb + kbase + (size_t)(kv0 + 32) * DMODEL,
                          lds_off(&Kl[nxt][0]), 64, 32, DMODEL);
        tdm_load_tile_f16(vt + vtbase + kv0 + 32,
                          lds_off(&Vt[nxt][0]), 32, 64, SEQ);
        __builtin_amdgcn_s_wait_tensorcnt(2);  // drain chunk it, keep it+1 flying
      } else {
        __builtin_amdgcn_s_wait_tensorcnt(0);
      }
    }
#else
#pragma unroll
    for (int t = 0; t < 16; ++t) {
      int idx = t * 128 + tid;
      int r = idx >> 6, c = idx & 63;
      Kl[cur][r * 64 + c] = kb[kbase + (size_t)(kv0 + r) * DMODEL + c];
      Vt[cur][c * 32 + r] = vt[vtbase + (size_t)c * SEQ + kv0 + r];
    }
#endif
    __syncthreads();

    // ---- scores 16x32 = (Q*scale) * K^T: Kl is N-major for K^T (ld=64) ----
    v8f s0 = vzero8(), s1 = vzero8();
    s0 = wmma_f16(aq0, frag_b_nmajor(&Kl[cur][0] + 0, 64), s0);
    s0 = wmma_f16(aq1, frag_b_nmajor(&Kl[cur][0] + 32, 64), s0);
    s1 = wmma_f16(aq0, frag_b_nmajor(&Kl[cur][0] + 16 * 64, 64), s1);
    s1 = wmma_f16(aq1, frag_b_nmajor(&Kl[cur][0] + 16 * 64 + 32, 64), s1);

    // ---- masked online softmax ----
    float sc8[8];
#pragma unroll
    for (int r = 0; r < 8; ++r) {
      const int qrow = q0 + hh * 8 + r;
      const unsigned char* mrow = mask + mb + (size_t)qrow * SEQ + kv0;
      float x0 = s0[r];
      float x1 = s1[r];
      if (mrow[n])      x0 = -1.0e9f;
      if (mrow[16 + n]) x1 = -1.0e9f;
      const float cm = redmax16(fmaxf(x0, x1));
      const float mn = fmaxf(m[r], cm);
      sc8[r] = __expf(m[r] - mn);
      m[r] = mn;
      const float p0 = __expf(x0 - mn);
      const float p1 = __expf(x1 - mn);
#pragma unroll
      for (int j = 0; j < 4; ++j) O[j][r] *= sc8[r];
      const int prow = hh * 8 + r;
      Plds[wid][prow * 32 + n]      = (_Float16)p0;
      Plds[wid][prow * 32 + 16 + n] = (_Float16)p1;
    }

    // ---- P fragment; row sums via WMMA vs ones; O += P * V ----
    const v16h ap = frag_a(&Plds[wid][0], 32);
    const v8f rowsum = wmma_f16(ap, ones, vzero8());
#pragma unroll
    for (int r = 0; r < 8; ++r) l[r] = l[r] * sc8[r] + rowsum[r];
#pragma unroll
    for (int j = 0; j < 4; ++j) {
      v16h bv = frag_b_nmajor(&Vt[cur][0] + (j * 16) * 32, 32);
      O[j] = wmma_f16(ap, bv, O[j]);
    }
    __syncthreads();
  }

#pragma unroll
  for (int j = 0; j < 4; ++j) {
#pragma unroll
    for (int r = 0; r < 8; ++r) {
      const int qrow = q0 + hh * 8 + r;
      const float inv = 1.0f / l[r];
      ctx[((size_t)b * SEQ + qrow) * DMODEL + h * DK + j * 16 + n] =
          (_Float16)(O[j][r] * inv);
    }
  }
}

// ------------------- row LayerNorm -------------------
__global__ __launch_bounds__(256) void layernorm_kernel(
    const float* __restrict__ x, const float* __restrict__ gamma,
    const float* __restrict__ beta, float* __restrict__ out) {
  __shared__ float ssum[8], ssq[8];
  const int row = blockIdx.x;
  const int tid = threadIdx.x;
  const float* xr = x + (size_t)row * DMODEL;
  float s = 0.0f, ss = 0.0f;
  for (int c = tid; c < DMODEL; c += 256) {
    float v = xr[c];
    s += v; ss += v * v;
  }
#pragma unroll
  for (int off = 16; off >= 1; off >>= 1) {
    s  += __shfl_xor(s,  off, 32);
    ss += __shfl_xor(ss, off, 32);
  }
  if ((tid & 31) == 0) { ssum[tid >> 5] = s; ssq[tid >> 5] = ss; }
  __syncthreads();
  float S = 0.0f, SS = 0.0f;
#pragma unroll
  for (int i = 0; i < 8; ++i) { S += ssum[i]; SS += ssq[i]; }
  const float mu   = S * (1.0f / DMODEL);
  const float var  = SS * (1.0f / DMODEL) - mu * mu;
  const float rstd = rsqrtf(var + 1e-5f);
  for (int c = tid; c < DMODEL; c += 256)
    out[(size_t)row * DMODEL + c] = (xr[c] - mu) * rstd * gamma[c] + beta[c];
}

extern "C" void kernel_launch(void* const* d_in, const int* in_sizes, int n_in,
                              void* d_out, int out_size, void* d_ws, size_t ws_size,
                              hipStream_t stream) {
  const float* Q  = (const float*)d_in[0];
  const float* K  = (const float*)d_in[1];
  const float* V  = (const float*)d_in[2];
  const unsigned char* mask = (const unsigned char*)d_in[3];
  const float* Wq = (const float*)d_in[4];
  const float* bq = (const float*)d_in[5];
  const float* Wk = (const float*)d_in[6];
  const float* bk = (const float*)d_in[7];
  const float* Wv = (const float*)d_in[8];
  const float* bv = (const float*)d_in[9];
  const float* Wo = (const float*)d_in[10];
  const float* bo = (const float*)d_in[11];
  const float* gamma = (const float*)d_in[12];
  const float* beta  = (const float*)d_in[13];
  float* out = (float*)d_out;

  char* ws = (char*)d_ws;
  const size_t hmat = (size_t)BS * DMODEL * sizeof(_Float16);       // 12.6 MB
  const size_t wmat = (size_t)DMODEL * DMODEL * sizeof(_Float16);   // 1.18 MB
  _Float16* Qx   = (_Float16*)(ws + 0 * hmat);   // f16 input copies
  _Float16* Kx   = (_Float16*)(ws + 1 * hmat);
  _Float16* Vx   = (_Float16*)(ws + 2 * hmat);
  _Float16* qh   = (_Float16*)(ws + 3 * hmat);   // projected q/k/v
  _Float16* kh   = (_Float16*)(ws + 4 * hmat);
  _Float16* vh   = (_Float16*)(ws + 5 * hmat);
  _Float16* ctxh = (_Float16*)(ws + 6 * hmat);
  _Float16* vt   = (_Float16*)(ws + 7 * hmat);   // per-head transposed V
  float*    projf = (float*)(ws + 8 * hmat);                        // 25.2 MB
  char*     wbase = ws + 8 * hmat + (size_t)BS * DMODEL * sizeof(float);
  _Float16* WqT = (_Float16*)(wbase + 0 * wmat);
  _Float16* WkT = (_Float16*)(wbase + 1 * wmat);
  _Float16* WvT = (_Float16*)(wbase + 2 * wmat);
  _Float16* WoT = (_Float16*)(wbase + 3 * wmat);

  const int an = BS * DMODEL;
  cvt_f16_kernel<<<an / 256, 256, 0, stream>>>(Q, Qx, an);
  cvt_f16_kernel<<<an / 256, 256, 0, stream>>>(K, Kx, an);
  cvt_f16_kernel<<<an / 256, 256, 0, stream>>>(V, Vx, an);
  const int wn = DMODEL * DMODEL;
  cvt_f16_transpose_kernel<<<wn / 256, 256, 0, stream>>>(Wq, WqT, DMODEL);
  cvt_f16_transpose_kernel<<<wn / 256, 256, 0, stream>>>(Wk, WkT, DMODEL);
  cvt_f16_transpose_kernel<<<wn / 256, 256, 0, stream>>>(Wv, WvT, DMODEL);
  cvt_f16_transpose_kernel<<<wn / 256, 256, 0, stream>>>(Wo, WoT, DMODEL);

  const dim3 gg(DMODEL / 64, BS / 128);
  gemm_wmma_kernel<0><<<gg, 256, 0, stream>>>(
      Qx, WqT, bq, nullptr, qh, nullptr, BS, DMODEL, DMODEL);
  gemm_wmma_kernel<0><<<gg, 256, 0, stream>>>(
      Kx, WkT, bk, nullptr, kh, nullptr, BS, DMODEL, DMODEL);
  gemm_wmma_kernel<0><<<gg, 256, 0, stream>>>(
      Vx, WvT, bv, nullptr, vh, nullptr, BS, DMODEL, DMODEL);
  transpose_v_kernel<<<dim3(SEQ / 64, NHEADS, BATCH), 256, 0, stream>>>(vh, vt);
  attn_kernel<<<dim3(SEQ / 64, NHEADS, BATCH), 128, 0, stream>>>(
      qh, kh, vt, mask, ctxh);
  gemm_wmma_kernel<1><<<gg, 256, 0, stream>>>(
      ctxh, WoT, bo, Q, nullptr, projf, BS, DMODEL, DMODEL);
  layernorm_kernel<<<BS, 256, 0, stream>>>(projf, gamma, beta, out);
}
